// Discriminator_IM_Sum_88064009437972
// MI455X (gfx1250) — matrix operand order repaired and checked by reading, not compiled, not run on hardware
//
#include <hip/hip_runtime.h>
#include <hip/hip_bf16.h>

// ---------------------------------------------------------------------------
// CDNA5 (gfx1250) implementation.
//  k_convert : f32 -> f16 copies of W_ih / W_hh (WMMA operands)
//  k_comb    : fold encoder (proj + concat + fusion) into one K=96 GEMM weight
//  k_encoder : enc = X @ Wc^T + b_eff     (WMMA f16, M=16384,N=256,K=96)
//  k_xproj   : xproj = enc @ W_ih0^T + b  (WMMA f16, M=16384,N=1024,K=256)
//  k_lstm    : 16384 sequential steps, 1 WGP, WMMA matvecs (h replicated in N),
//              TDM (tensor_load_to_lds) double-buffers next xproj row into LDS
//  k_fc      : fc1+relu, fc2+sigmoid on last 64 hidden rows
// ---------------------------------------------------------------------------

typedef __attribute__((ext_vector_type(16))) _Float16 v16h;
typedef __attribute__((ext_vector_type(8)))  _Float16 v8h;
typedef __attribute__((ext_vector_type(8)))  float    v8f;
typedef __attribute__((ext_vector_type(4)))  unsigned int u32x4;
typedef __attribute__((ext_vector_type(8)))  int i32x8;
typedef __attribute__((ext_vector_type(4)))  int i32x4;

#define WMMA_F16F32(a, b, c) \
  __builtin_amdgcn_wmma_f32_16x16x32_f16(false, (a), false, (b), (short)0, (c), false, false)

// A-matrix 16x32 f16 fragment (ISA 7.12.2): lane m=lane%16,
// elems j0..7 -> k = kbase + halfoff + j, j8..15 -> k = kbase + 16 + halfoff + j
__device__ __forceinline__ v16h frag_a(const _Float16* rowp, int kbase, int half) {
  v16h a;
  const _Float16* p = rowp + kbase + (half ? 8 : 0);
  ((v8h*)&a)[0] = *(const v8h*)(p);
  ((v8h*)&a)[1] = *(const v8h*)(p + 16);
  return a;
}
// B-matrix 32x16 f16 fragment: lane n=lane%16, same k order as frag_a
__device__ __forceinline__ v16h frag_b(const _Float16* rowp, int kbase, int half) {
  v16h b;
  const _Float16* p = rowp + kbase + (half ? 8 : 0);
  ((v8h*)&b)[0] = *(const v8h*)(p);
  ((v8h*)&b)[1] = *(const v8h*)(p + 16);
  return b;
}

__device__ __forceinline__ float sigmoidf_(float x) { return 1.0f / (1.0f + __expf(-x)); }

// TDM: DMA one 4KB row (1024 f32) from global memory into LDS.
// D# group0: count=1, lds_addr, 57-bit global_addr, type=2 ("image").
// D# group1: data_size=4B, tensor 1024x1, tile_dim0=1024 (1D), strides=1024.
__device__ __forceinline__ void tdm_load_row(const float* gaddr, unsigned lds_byte_addr) {
  unsigned long long ga = (unsigned long long)gaddr;
  u32x4 g0;
  g0[0] = 1u;                                                   // count=1, user mode
  g0[1] = lds_byte_addr;                                        // LDS dest (bytes)
  g0[2] = (unsigned)ga;                                         // global_addr[31:0]
  g0[3] = (unsigned)((ga >> 32) & 0x01FFFFFFull) | (2u << 30);  // ga[56:32] | type=2
  i32x8 g1;
  g1[0] = 0x00020000;          // wg_mask=0 | data_size=2 (4B) | no barrier/iterate/pad
  g1[1] = (int)(1024u << 16);  // atomic_barrier_addr=0 | tensor_dim0[15:0]=1024
  g1[2] = (int)(1u << 16);     // tensor_dim0[31:16]=0 | tensor_dim1[15:0]=1
  g1[3] = (int)(1024u << 16);  // tensor_dim1[31:16]=0 | tile_dim0=1024
  g1[4] = 0;                   // tile_dim1=0 (unused), tile_dim2=0
  g1[5] = 1024;                // tensor_dim0_stride[31:0]
  g1[6] = (int)(1024u << 16);  // stride0[47:32]=0 | tensor_dim1_stride[15:0]
  g1[7] = 0;                   // tensor_dim1_stride[47:16]=0
  i32x4 z4 = {0, 0, 0, 0};
#if __clang_major__ >= 23
  i32x8 z8 = {0, 0, 0, 0, 0, 0, 0, 0};
  __builtin_amdgcn_tensor_load_to_lds(g0, g1, z4, z4, z8, 0);
#else
  __builtin_amdgcn_tensor_load_to_lds(g0, g1, z4, z4, 0);
#endif
}

// ---------------------------------------------------------------------------
__global__ void k_convert(const float* __restrict__ Wih, const float* __restrict__ Whh,
                          _Float16* __restrict__ Wih16, _Float16* __restrict__ Whh16) {
  const int n = 3 * 1024 * 256;
  for (int i = blockIdx.x * blockDim.x + threadIdx.x; i < n; i += gridDim.x * blockDim.x) {
    Wih16[i] = (_Float16)Wih[i];
    Whh16[i] = (_Float16)Whh[i];
  }
}

// Fold encoder: Wc[n][k<25] = sum_f Wfus[n,f]*Wemo[f,k];  k in [25,83): 3dmm part; pad->0.
// b_eff[n] = b_fus[n] + 2*Wfus[n,:256]@b_emo + 2*Wfus[n,256:]@b_3d
__global__ __launch_bounds__(128) void k_comb(
    const float* __restrict__ W_emo, const float* __restrict__ b_emo,
    const float* __restrict__ W_3d,  const float* __restrict__ b_3d,
    const float* __restrict__ W_fus, const float* __restrict__ b_fus,
    _Float16* __restrict__ Wc, float* __restrict__ b_eff) {
  const int n = blockIdx.x;      // 0..255
  const int k = threadIdx.x;     // 0..127
  if (k < 96) {
    float acc = 0.0f;
    if (k < 25) {
      for (int f = 0; f < 256; ++f) acc += W_fus[n * 512 + f] * W_emo[f * 25 + k];
    } else if (k < 83) {
      const int k2 = k - 25;
      for (int f = 0; f < 256; ++f) acc += W_fus[n * 512 + 256 + f] * W_3d[f * 58 + k2];
    }
    Wc[n * 96 + k] = (_Float16)acc;
  } else if (k == 96) {
    float acc = b_fus[n];
    for (int f = 0; f < 256; ++f) acc += 2.0f * W_fus[n * 512 + f] * b_emo[f];
    for (int f = 0; f < 256; ++f) acc += 2.0f * W_fus[n * 512 + 256 + f] * b_3d[f];
    b_eff[n] = acc;
  }
}

// enc16[s][n], s = t*64 + b  (xs row order of the reference scan)
__global__ __launch_bounds__(256) void k_encoder(
    const float* __restrict__ le, const float* __restrict__ se,
    const float* __restrict__ l3, const float* __restrict__ s3,
    const _Float16* __restrict__ Wc, const float* __restrict__ b_eff,
    _Float16* __restrict__ enc16) {
  __shared__ __align__(16) _Float16 At[16][96];
  const int s0 = blockIdx.x * 16;
  const int tid = threadIdx.x;
  for (int i = tid; i < 16 * 96; i += 256) {
    const int r = i / 96, k = i % 96;
    const int s = s0 + r;
    const int t = s >> 6, bb = s & 63, sb = bb >> 2;
    float v = 0.0f;
    if (k < 25)      v = le[(bb * 256 + t) * 25 + k]        + se[(sb * 256 + t) * 25 + k];
    else if (k < 83) v = l3[(bb * 256 + t) * 58 + (k - 25)] + s3[(sb * 256 + t) * 58 + (k - 25)];
    At[r][k] = (_Float16)v;
  }
  __syncthreads();
  const int wave = tid >> 5, lane = tid & 31;
  const int half = lane >> 4, lm = lane & 15;
  for (int nt = wave; nt < 16; nt += 8) {
    v8f acc = {};
    const _Float16* brow = Wc + (nt * 16 + lm) * 96;
    for (int kc = 0; kc < 3; ++kc) {
      v16h af = frag_a(&At[lm][0], kc * 32, half);
      v16h bf = frag_b(brow, kc * 32, half);
      acc = WMMA_F16F32(af, bf, acc);
    }
    const int n = nt * 16 + lm;
    const float bias = b_eff[n];
    for (int r = 0; r < 8; ++r) {
      const int m = r + half * 8;
      enc16[(size_t)(s0 + m) * 256 + n] = (_Float16)(acc[r] + bias);
    }
  }
}

// xproj[s][g] = enc[s] @ W_ih0^T + b_ih0  (the non-recurrent LSTM work, hoisted)
__global__ __launch_bounds__(256) void k_xproj(
    const _Float16* __restrict__ enc16, const _Float16* __restrict__ Wih0,
    const float* __restrict__ b_ih0, float* __restrict__ xproj) {
  const int wave = threadIdx.x >> 5, lane = threadIdx.x & 31;
  const int half = lane >> 4, lm = lane & 15;
  const int flat = blockIdx.x * 8 + wave;     // 65536 wave-tiles
  const int mt = flat >> 6, nt = flat & 63;
  const _Float16* arow = enc16 + (size_t)(mt * 16 + lm) * 256;
  const _Float16* brow = Wih0 + (size_t)(nt * 16 + lm) * 256;
  v8f acc = {};
  for (int kc = 0; kc < 8; ++kc) {
    v16h af = frag_a(arow, kc * 32, half);
    v16h bf = frag_b(brow, kc * 32, half);
    acc = WMMA_F16F32(af, bf, acc);
  }
  const int n = nt * 16 + lm;
  const float bias = b_ih0[n];
  for (int r = 0; r < 8; ++r) {
    const int m = mt * 16 + r + half * 8;
    xproj[(size_t)m * 1024 + n] = acc[r] + bias;
  }
}

// Sequential 3-layer LSTM scan: one persistent workgroup (32 wave32 = 1 WGP).
// Recurrent matvecs via WMMA (h replicated across B columns); next step's
// xproj row is DMA'd into an LDS double buffer by the Tensor Data Mover.
__global__ __launch_bounds__(1024) void k_lstm(
    const _Float16* __restrict__ Wih16, const _Float16* __restrict__ Whh16,
    const float* __restrict__ b_ih, const float* __restrict__ b_hh,
    const float* __restrict__ xproj, float* __restrict__ ys_tail) {
  __shared__ __align__(16) _Float16 h16[3][256];
  __shared__ float cst[3][256];
  __shared__ float gbuf[1024];
  __shared__ float biasL[3][1024];
  __shared__ __align__(16) float xpbuf[2][1024];   // TDM double buffer

  const int tid = threadIdx.x;
  biasL[0][tid] = b_hh[tid];                               // b_ih0 folded into xproj
  biasL[1][tid] = b_ih[1024 + tid] + b_hh[1024 + tid];
  biasL[2][tid] = b_ih[2048 + tid] + b_hh[2048 + tid];
  if (tid < 256) {
    for (int l = 0; l < 3; ++l) { h16[l][tid] = (_Float16)0.0f; cst[l][tid] = 0.0f; }
  }

  const int wave = tid >> 5, lane = tid & 31;
  const int half = lane >> 4, lm = lane & 15;
  const unsigned xp_lds0 = (unsigned)(unsigned long long)&xpbuf[0][0];
  const unsigned xp_lds1 = (unsigned)(unsigned long long)&xpbuf[1][0];

  if (wave == 0) tdm_load_row(xproj, xp_lds0);   // prefetch row 0
  __syncthreads();

  const _Float16* Whh0 = Whh16;
  const _Float16* Wih1 = Wih16 + 1024 * 256;
  const _Float16* Whh1 = Whh16 + 1024 * 256;
  const _Float16* Wih2 = Wih16 + 2 * 1024 * 256;
  const _Float16* Whh2 = Whh16 + 2 * 1024 * 256;

  for (int s = 0; s < 16384; ++s) {
    // -------- layer 0: g = xproj[s] + Whh0 @ h0 + b --------
    for (int tt = wave; tt < 64; tt += 32) {
      v8f acc = {};
      const _Float16* arow = Whh0 + (tt * 16 + lm) * 256;
      for (int kc = 0; kc < 8; ++kc)
        acc = WMMA_F16F32(frag_a(arow, kc * 32, half), frag_b(&h16[0][0], kc * 32, half), acc);
      if (lm == 0) {
        float* gp = &gbuf[tt * 16 + half * 8];
        for (int r = 0; r < 8; ++r) gp[r] = acc[r];   // column 0 of D
      }
    }
    if (wave == 0) __builtin_amdgcn_s_wait_tensorcnt(0);  // xpbuf[s&1] ready
    __syncthreads();
    if (tid < 256) {
      const float* xp = &xpbuf[s & 1][0];
      const float gi = xp[tid]       + gbuf[tid]       + biasL[0][tid];
      const float gf = xp[256 + tid] + gbuf[256 + tid] + biasL[0][256 + tid];
      const float gg = xp[512 + tid] + gbuf[512 + tid] + biasL[0][512 + tid];
      const float go = xp[768 + tid] + gbuf[768 + tid] + biasL[0][768 + tid];
      const float c = sigmoidf_(gf) * cst[0][tid] + sigmoidf_(gi) * tanhf(gg);
      const float h = sigmoidf_(go) * tanhf(c);
      cst[0][tid] = c; h16[0][tid] = (_Float16)h;
    }
    __syncthreads();
    // kick off DMA of next step's xproj row (overlaps layers 1-2 + next layer 0)
    if (wave == 0 && s + 1 < 16384)
      tdm_load_row(xproj + (size_t)(s + 1) * 1024, ((s + 1) & 1) ? xp_lds1 : xp_lds0);
    // -------- layer 1: g = Wih1 @ h0 + Whh1 @ h1 + b --------
    for (int tt = wave; tt < 64; tt += 32) {
      v8f acc = {};
      const _Float16* ar0 = Wih1 + (tt * 16 + lm) * 256;
      const _Float16* ar1 = Whh1 + (tt * 16 + lm) * 256;
      for (int kc = 0; kc < 8; ++kc) {
        acc = WMMA_F16F32(frag_a(ar0, kc * 32, half), frag_b(&h16[0][0], kc * 32, half), acc);
        acc = WMMA_F16F32(frag_a(ar1, kc * 32, half), frag_b(&h16[1][0], kc * 32, half), acc);
      }
      if (lm == 0) {
        float* gp = &gbuf[tt * 16 + half * 8];
        for (int r = 0; r < 8; ++r) gp[r] = acc[r];
      }
    }
    __syncthreads();
    if (tid < 256) {
      const float gi = gbuf[tid]       + biasL[1][tid];
      const float gf = gbuf[256 + tid] + biasL[1][256 + tid];
      const float gg = gbuf[512 + tid] + biasL[1][512 + tid];
      const float go = gbuf[768 + tid] + biasL[1][768 + tid];
      const float c = sigmoidf_(gf) * cst[1][tid] + sigmoidf_(gi) * tanhf(gg);
      const float h = sigmoidf_(go) * tanhf(c);
      cst[1][tid] = c; h16[1][tid] = (_Float16)h;
    }
    __syncthreads();
    // -------- layer 2: g = Wih2 @ h1 + Whh2 @ h2 + b --------
    for (int tt = wave; tt < 64; tt += 32) {
      v8f acc = {};
      const _Float16* ar0 = Wih2 + (tt * 16 + lm) * 256;
      const _Float16* ar1 = Whh2 + (tt * 16 + lm) * 256;
      for (int kc = 0; kc < 8; ++kc) {
        acc = WMMA_F16F32(frag_a(ar0, kc * 32, half), frag_b(&h16[1][0], kc * 32, half), acc);
        acc = WMMA_F16F32(frag_a(ar1, kc * 32, half), frag_b(&h16[2][0], kc * 32, half), acc);
      }
      if (lm == 0) {
        float* gp = &gbuf[tt * 16 + half * 8];
        for (int r = 0; r < 8; ++r) gp[r] = acc[r];
      }
    }
    __syncthreads();
    if (tid < 256) {
      const float gi = gbuf[tid]       + biasL[2][tid];
      const float gf = gbuf[256 + tid] + biasL[2][256 + tid];
      const float gg = gbuf[512 + tid] + biasL[2][512 + tid];
      const float go = gbuf[768 + tid] + biasL[2][768 + tid];
      const float c = sigmoidf_(gf) * cst[2][tid] + sigmoidf_(gi) * tanhf(gg);
      const float h = sigmoidf_(go) * tanhf(c);
      cst[2][tid] = c; h16[2][tid] = (_Float16)h;
      if (s >= 16384 - 64) ys_tail[(size_t)(s - (16384 - 64)) * 256 + tid] = h;  // ys[-B:]
    }
    __syncthreads();
  }
}

// out[i] = sigmoid( relu(h2 @ Wfc1^T + b1) @ Wfc2^T + b2 )
__global__ __launch_bounds__(256) void k_fc(
    const float* __restrict__ ys_tail,
    const float* __restrict__ W_fc1, const float* __restrict__ b_fc1,
    const float* __restrict__ W_fc2, const float* __restrict__ b_fc2,
    float* __restrict__ out) {
  __shared__ float hrow[256];
  __shared__ float red[256];
  const int row = blockIdx.x, tid = threadIdx.x;
  hrow[tid] = ys_tail[(size_t)row * 256 + tid];
  __syncthreads();
  float acc = b_fc1[tid];
  for (int k = 0; k < 256; ++k) acc += W_fc1[tid * 256 + k] * hrow[k];
  red[tid] = fmaxf(acc, 0.0f) * W_fc2[tid];
  __syncthreads();
  for (int off = 128; off > 0; off >>= 1) {
    if (tid < off) red[tid] += red[tid + off];
    __syncthreads();
  }
  if (tid == 0) out[row] = sigmoidf_(red[0] + b_fc2[0]);
}

// ---------------------------------------------------------------------------
extern "C" void kernel_launch(void* const* d_in, const int* in_sizes, int n_in,
                              void* d_out, int out_size, void* d_ws, size_t ws_size,
                              hipStream_t stream) {
  (void)in_sizes; (void)n_in; (void)out_size; (void)ws_size;
  const float* se   = (const float*)d_in[0];
  const float* le   = (const float*)d_in[1];
  const float* s3   = (const float*)d_in[2];
  const float* l3   = (const float*)d_in[3];
  const float* Wemo = (const float*)d_in[4];
  const float* bemo = (const float*)d_in[5];
  const float* W3d  = (const float*)d_in[6];
  const float* b3d  = (const float*)d_in[7];
  const float* Wfus = (const float*)d_in[8];
  const float* bfus = (const float*)d_in[9];
  const float* Wih  = (const float*)d_in[10];
  const float* Whh  = (const float*)d_in[11];
  const float* bih  = (const float*)d_in[12];
  const float* bhh  = (const float*)d_in[13];
  const float* Wfc1 = (const float*)d_in[14];
  const float* bfc1 = (const float*)d_in[15];
  const float* Wfc2 = (const float*)d_in[16];
  const float* bfc2 = (const float*)d_in[17];
  float* out = (float*)d_out;

  char* ws = (char*)d_ws;
  _Float16* enc16   = (_Float16*)(ws);                    // 16384*256*2  = 8,388,608
  float*    xproj   = (float*)   (ws + 8388608);          // 16384*1024*4 = 67,108,864
  _Float16* Wc      = (_Float16*)(ws + 75497472);         // 256*96*2     = 49,152
  float*    b_eff   = (float*)   (ws + 75546624);         // 256*4        = 1,024
  _Float16* Wih16   = (_Float16*)(ws + 75547648);         // 3*1024*256*2 = 1,572,864
  _Float16* Whh16   = (_Float16*)(ws + 77120512);         // 3*1024*256*2 = 1,572,864
  float*    ys_tail = (float*)   (ws + 78693376);         // 64*256*4     = 65,536

  k_convert<<<512, 256, 0, stream>>>(Wih, Whh, Wih16, Whh16);
  k_comb<<<256, 128, 0, stream>>>(Wemo, bemo, W3d, b3d, Wfus, bfus, Wc, b_eff);
  k_encoder<<<1024, 256, 0, stream>>>(le, se, l3, s3, Wc, b_eff, enc16);
  k_xproj<<<8192, 256, 0, stream>>>(enc16, Wih16, bih, xproj);
  k_lstm<<<1, 1024, 0, stream>>>(Wih16, Whh16, bih, bhh, xproj, ys_tail);
  k_fc<<<64, 256, 0, stream>>>(ys_tail, Wfc1, bfc1, Wfc2, bfc2, out);
}
